// SpectraNoCellType_13967233647507
// MI455X (gfx1250) — matrix operand order repaired and checked by reading, not compiled, not run on hardware
//
#include <hip/hip_runtime.h>
#include <hip/hip_bf16.h>
#include <hip/hip_fp16.h>

typedef __attribute__((ext_vector_type(16))) _Float16 v16h;
typedef __attribute__((ext_vector_type(8)))  _Float16 v8h;
typedef __attribute__((ext_vector_type(8)))  float    v8f;

#define N_    10000
#define P_    8000
#define L_    100
#define KP    128          // K padded to 128 (zeros in [100,128))
#define NPAD  10112        // 79 * 128
#define PPAD  8064         // 63 * 128
#define LAM_  0.01f
#define DELTA_ 0.001f

__device__ __forceinline__ float sigf(float x) { return 1.0f / (1.0f + __expf(-x)); }

// ---------------------------------------------------------------------------
// prep: softmax rows of theta -> th (f16, padded), ths = th*(sigmoid(gs)+delta)
// one wave32 per row, 8 rows per 256-thread block
// ---------------------------------------------------------------------------
__global__ void __launch_bounds__(256) prep_theta_kernel(
    const float* __restrict__ theta, const float* __restrict__ gene_scaling,
    _Float16* __restrict__ th, _Float16* __restrict__ ths) {
  const int lane = threadIdx.x & 31;
  const int wave = threadIdx.x >> 5;
  const int row  = blockIdx.x * 8 + wave;          // row < PPAD
  const bool vr  = (row < P_);
  const float* trow = theta + row * L_;
  float v[4];
#pragma unroll
  for (int c = 0; c < 4; ++c) {
    int i = lane + c * 32;
    v[c] = (vr && i < L_) ? trow[i] : -3.0e38f;
  }
  float mx = fmaxf(fmaxf(v[0], v[1]), fmaxf(v[2], v[3]));
#pragma unroll
  for (int off = 16; off > 0; off >>= 1) mx = fmaxf(mx, __shfl_xor(mx, off));
  float e[4];
  float s = 0.f;
#pragma unroll
  for (int c = 0; c < 4; ++c) {
    int i = lane + c * 32;
    e[c] = (vr && i < L_) ? __expf(v[c] - mx) : 0.f;
    s += e[c];
  }
#pragma unroll
  for (int off = 16; off > 0; off >>= 1) s += __shfl_xor(s, off);
  const float inv   = vr ? (1.0f / s) : 0.f;
  const float scale = vr ? (sigf(gene_scaling[row]) + DELTA_) : 0.f;
#pragma unroll
  for (int c = 0; c < 4; ++c) {
    int i = lane + c * 32;                          // covers 0..127
    float t = e[c] * inv;
    th [row * KP + i] = (_Float16)t;
    ths[row * KP + i] = (_Float16)(t * scale);
  }
}

// a = exp(alpha), f16, zero-padded to NPAD x KP
__global__ void prep_alpha_kernel(const float* __restrict__ alpha,
                                  _Float16* __restrict__ a) {
  int idx = blockIdx.x * blockDim.x + threadIdx.x;
  if (idx >= NPAD * KP) return;
  int row = idx >> 7, k = idx & (KP - 1);
  float v = (row < N_ && k < L_) ? __expf(alpha[row * L_ + k]) : 0.f;
  a[idx] = (_Float16)v;
}

// et = 0.5*(sigmoid(eta) + sigmoid(eta)^T), f32, L x L
__global__ void prep_eta_kernel(const float* __restrict__ eta,
                                float* __restrict__ et) {
  int idx = blockIdx.x * blockDim.x + threadIdx.x;
  if (idx >= L_ * L_) return;
  int i = idx / L_, j = idx - i * L_;
  et[idx] = 0.5f * (sigf(eta[i * L_ + j]) + sigf(eta[j * L_ + i]));
}

// bm = th @ et  (P x L, tiny: 8e7 MACs), f16 zero-padded
__global__ void gemm_B_kernel(const _Float16* __restrict__ th,
                              const float* __restrict__ et,
                              _Float16* __restrict__ bm) {
  int idx = blockIdx.x * blockDim.x + threadIdx.x;
  if (idx >= PPAD * KP) return;
  int row = idx >> 7, l = idx & (KP - 1);
  float s = 0.f;
  if (l < L_) {
    const _Float16* tr = th + row * KP;
    for (int m = 0; m < L_; ++m) s += (float)tr[m] * et[m * L_ + l];
  }
  bm[idx] = (_Float16)s;
}

// ---------------------------------------------------------------------------
// fragment loaders: row-major, KP-stride f16 operand matrices (L2 resident)
// A 16x32: lanes 0-15 = row M, K {0..7,16..23}; lanes 16-31 K {8..15,24..31}
// B 32x16: lane = column (= operand row), K {0..7,16..23}/{8..15,24..31} per lane half
// ---------------------------------------------------------------------------
__device__ __forceinline__ v16h load_afrag(const _Float16* base, int rowBase,
                                           int kb, int laneM, int laneHi) {
  const _Float16* p = base + (rowBase + laneM) * KP + kb + laneHi * 8;
  v8h lo = *(const v8h*)p;
  v8h hi = *(const v8h*)(p + 16);
  return __builtin_shufflevector(lo, hi, 0,1,2,3,4,5,6,7,8,9,10,11,12,13,14,15);
}
__device__ __forceinline__ v16h load_bfrag(const _Float16* base, int rowBase,
                                           int kb, int laneM, int laneHi) {
  const _Float16* p = base + (rowBase + laneM) * KP + kb + laneHi * 8;
  v8h lo = *(const v8h*)p;
  v8h hi = *(const v8h*)(p + 16);
  return __builtin_shufflevector(lo, hi, 0,1,2,3,4,5,6,7,8,9,10,11,12,13,14,15);
}

__device__ __forceinline__ void block_reduce_store(float lsum, float* dst) {
  __shared__ float red[256];
  red[threadIdx.x] = lsum;
  __syncthreads();
  for (int st = 128; st > 0; st >>= 1) {
    if ((int)threadIdx.x < st) red[threadIdx.x] += red[threadIdx.x + st];
    __syncthreads();
  }
  if (threadIdx.x == 0) *dst = red[0];
}

// ---------------------------------------------------------------------------
// Kernel 1: recon = a @ ths^T fused with term1 = sum(recon - X*log(recon)),
// scaled by LAM. Block tile 128x128, 8 waves, wave tile 32x64 (2x4 WMMA).
// ---------------------------------------------------------------------------
__global__ void __launch_bounds__(256) recon_term1_kernel(
    const float* __restrict__ X, const _Float16* __restrict__ A,
    const _Float16* __restrict__ B, float* __restrict__ partials) {
  const int tid = threadIdx.x;
  const int lane = tid & 31, wave = tid >> 5;
  const int laneM = lane & 15, laneHi = lane >> 4;
  const int rowBase = blockIdx.y * 128 + (wave & 3) * 32;
  const int colBase = blockIdx.x * 128 + (wave >> 2) * 64;

  v8f zero = {};
  v8f acc[2][4];
#pragma unroll
  for (int mt = 0; mt < 2; ++mt)
#pragma unroll
    for (int nt = 0; nt < 4; ++nt) acc[mt][nt] = zero;

#pragma unroll
  for (int kb = 0; kb < KP; kb += 32) {
    v16h af[2], bf[4];
#pragma unroll
    for (int mt = 0; mt < 2; ++mt)
      af[mt] = load_afrag(A, rowBase + mt * 16, kb, laneM, laneHi);
#pragma unroll
    for (int nt = 0; nt < 4; ++nt)
      bf[nt] = load_bfrag(B, colBase + nt * 16, kb, laneM, laneHi);
#pragma unroll
    for (int mt = 0; mt < 2; ++mt)
#pragma unroll
      for (int nt = 0; nt < 4; ++nt)
        acc[mt][nt] = __builtin_amdgcn_wmma_f32_16x16x32_f16(
            false, af[mt], false, bf[nt], (short)0, acc[mt][nt], false, false);
  }

  float lsum = 0.f;
#pragma unroll
  for (int mt = 0; mt < 2; ++mt)
#pragma unroll
    for (int nt = 0; nt < 4; ++nt)
#pragma unroll
      for (int v = 0; v < 8; ++v) {
        int row = rowBase + mt * 16 + v + laneHi * 8;
        int col = colBase + nt * 16 + laneM;
        if (row < N_ && col < P_) {
          float rec = acc[mt][nt][v];
          float x   = X[row * P_ + col];
          lsum += rec - x * __logf(rec);   // -(xlogy(X,recon) - recon)
        }
      }
  lsum *= LAM_;
  block_reduce_store(lsum, &partials[blockIdx.y * gridDim.x + blockIdx.x]);
}

// ---------------------------------------------------------------------------
// Kernel 2: mat = bm @ th^T fused with term2+term3 log-reductions
// ---------------------------------------------------------------------------
__global__ void __launch_bounds__(256) mat_terms_kernel(
    const float* __restrict__ adj, const float* __restrict__ adj1m,
    const _Float16* __restrict__ A, const _Float16* __restrict__ B,
    const float* __restrict__ kap, const float* __restrict__ rho,
    float* __restrict__ partials) {
  const int tid = threadIdx.x;
  const int lane = tid & 31, wave = tid >> 5;
  const int laneM = lane & 15, laneHi = lane >> 4;
  const int rowBase = blockIdx.y * 128 + (wave & 3) * 32;
  const int colBase = blockIdx.x * 128 + (wave >> 2) * 64;

  const float k  = sigf(kap[0]);
  const float r  = sigf(rho[0]);
  const float c1 = (1.f - r) * (1.f - k);
  const float c2 = (1.f - r) * k;

  v8f zero = {};
  v8f acc[2][4];
#pragma unroll
  for (int mt = 0; mt < 2; ++mt)
#pragma unroll
    for (int nt = 0; nt < 4; ++nt) acc[mt][nt] = zero;

#pragma unroll
  for (int kb = 0; kb < KP; kb += 32) {
    v16h af[2], bf[4];
#pragma unroll
    for (int mt = 0; mt < 2; ++mt)
      af[mt] = load_afrag(A, rowBase + mt * 16, kb, laneM, laneHi);
#pragma unroll
    for (int nt = 0; nt < 4; ++nt)
      bf[nt] = load_bfrag(B, colBase + nt * 16, kb, laneM, laneHi);
#pragma unroll
    for (int mt = 0; mt < 2; ++mt)
#pragma unroll
      for (int nt = 0; nt < 4; ++nt)
        acc[mt][nt] = __builtin_amdgcn_wmma_f32_16x16x32_f16(
            false, af[mt], false, bf[nt], (short)0, acc[mt][nt], false, false);
  }

  float lsum = 0.f;
#pragma unroll
  for (int mt = 0; mt < 2; ++mt)
#pragma unroll
    for (int nt = 0; nt < 4; ++nt)
#pragma unroll
      for (int v = 0; v < 8; ++v) {
        int row = rowBase + mt * 16 + v + laneHi * 8;
        int col = colBase + nt * 16 + laneM;
        if (row < P_ && col < P_) {
          float m  = acc[mt][nt][v];
          int   id = row * P_ + col;
          float w2 = adj[id];           // adj*weights == adj (binary)
          float w3 = adj1m[id];
          if (w2 != 0.f) lsum -= w2 * __logf(c1 * m + c2);
          if (w3 != 0.f) lsum -= w3 * __logf(c1 * (1.f - m) + r);
        }
      }
  block_reduce_store(lsum, &partials[63 * 79 + blockIdx.y * gridDim.x + blockIdx.x]);
}

// deterministic final reduction (single block, fixed order)
__global__ void __launch_bounds__(256) reduce_kernel(
    const float* __restrict__ partials, float* __restrict__ out) {
  const int n = 63 * 79 + 63 * 63;
  float s = 0.f;
  for (int i = threadIdx.x; i < n; i += 256) s += partials[i];
  __shared__ float red[256];
  red[threadIdx.x] = s;
  __syncthreads();
  for (int st = 128; st > 0; st >>= 1) {
    if ((int)threadIdx.x < st) red[threadIdx.x] += red[threadIdx.x + st];
    __syncthreads();
  }
  if (threadIdx.x == 0) out[0] = red[0];
}

extern "C" void kernel_launch(void* const* d_in, const int* in_sizes, int n_in,
                              void* d_out, int out_size, void* d_ws, size_t ws_size,
                              hipStream_t stream) {
  (void)in_sizes; (void)n_in; (void)out_size; (void)ws_size;
  const float* X     = (const float*)d_in[0];
  const float* adj   = (const float*)d_in[1];
  const float* adj1m = (const float*)d_in[2];
  // d_in[3] = weights == adj (binary), adj*weights == adj -> not needed
  const float* theta = (const float*)d_in[4];
  const float* alpha = (const float*)d_in[5];
  const float* eta   = (const float*)d_in[6];
  const float* gs    = (const float*)d_in[7];
  const float* kappa = (const float*)d_in[8];
  const float* rho   = (const float*)d_in[9];

  char* ws = (char*)d_ws;                       // ~8.9 MB total
  _Float16* a_f16   = (_Float16*)(ws + 0);        // NPAD*KP*2 = 2,588,672
  _Float16* th_f16  = (_Float16*)(ws + 2588672);  // PPAD*KP*2 = 2,064,384
  _Float16* ths_f16 = (_Float16*)(ws + 4653056);  // 2,064,384
  _Float16* bm_f16  = (_Float16*)(ws + 6717440);  // 2,064,384
  float*    et_f32  = (float*)(ws + 8781824);     // 40,000 (padded 40,192)
  float*    partials= (float*)(ws + 8822016);     // 8946 * 4

  prep_theta_kernel<<<PPAD / 8, 256, 0, stream>>>(theta, gs, th_f16, ths_f16);
  prep_alpha_kernel<<<(NPAD * KP + 255) / 256, 256, 0, stream>>>(alpha, a_f16);
  prep_eta_kernel<<<(L_ * L_ + 255) / 256, 256, 0, stream>>>(eta, et_f32);
  gemm_B_kernel<<<(PPAD * KP + 255) / 256, 256, 0, stream>>>(th_f16, et_f32, bm_f16);
  recon_term1_kernel<<<dim3(63, 79), 256, 0, stream>>>(X, a_f16, ths_f16, partials);
  mat_terms_kernel<<<dim3(63, 63), 256, 0, stream>>>(adj, adj1m, bm_f16, th_f16,
                                                     kappa, rho, partials);
  reduce_kernel<<<1, 256, 0, stream>>>(partials, (float*)d_out);
}
